// DiffiT_51075751084662
// MI455X (gfx1250) — matrix operand order repaired and checked by reading, not compiled, not run on hardware
//
#include <hip/hip_runtime.h>
#include <hip/hip_bf16.h>

typedef _Float16 v16h __attribute__((ext_vector_type(16)));
typedef _Float16 v8h  __attribute__((ext_vector_type(8)));
typedef float    v8f  __attribute__((ext_vector_type(8)));

#define BB 16
#define NN 1024
#define CC 1152
#define HH 16
#define HD 72
#define HDP 96
#define HID 3072
#define MM (BB*NN)
#define C3 (3*CC)

// ---------------------------------------------------------------------------
// WMMA helpers (CDNA5 f16 16x16x32, f32 accumulate)
// ---------------------------------------------------------------------------
__device__ __forceinline__ v8f wmma16(v16h a, v16h b, v8f c) {
  // 8 args: (neg_a, A, neg_b, B, c_mod, C, reuse_a, reuse_b)
  return __builtin_amdgcn_wmma_f32_16x16x32_f16(false, a, false, b, (short)0, c,
                                                false, false);
}

// A-matrix 16x32 f16 fragment (ISA layout: lane<16 holds K[0..7],[16..23],
// lane>=16 holds K[8..15],[24..31] of row M=lane&15) -> two 16B loads.
__device__ __forceinline__ v16h load_a_frag(const _Float16* __restrict__ A,
                                            int lda, int m0, int k0, int lane) {
  const int r = lane & 15, hl = lane >> 4;
  const _Float16* p = A + (size_t)(m0 + r) * lda + k0 + hl * 8;
  v16h a;
  ((v8h*)&a)[0] = *(const v8h*)(p);
  ((v8h*)&a)[1] = *(const v8h*)(p + 16);
  return a;
}

// B-matrix 32x16 f16 fragment: lane n holds K=k0+hl*8..+8 and k0+16+hl*8..+8
// of output column n0+n (same K order as the A fragment) -> two 16B loads.
__device__ __forceinline__ v16h load_b_frag(const _Float16* __restrict__ W,
                                            int ldw, int n0, int k0, int lane) {
  const int r = lane & 15, hl = lane >> 4;
  const _Float16* p = W + (size_t)(n0 + r) * ldw + k0 + hl * 8;
  v16h b;
  ((v8h*)&b)[0] = *(const v8h*)(p);
  ((v8h*)&b)[1] = *(const v8h*)(p + 16);
  return b;
}

// Async global -> LDS 16B copy (VFLAT async op, tracked by ASYNCcnt).
// VDST VGPR carries the LDS byte address (low 32 bits of the flat address),
// VADDR the 64-bit global address.
__device__ __forceinline__ void async_g2l_b128(void* lds, const _Float16* g) {
  asm volatile("global_load_async_to_lds_b128 %0, %1, off"
               :: "v"((unsigned)(unsigned long long)(uintptr_t)lds),
                  "v"((unsigned long long)(uintptr_t)g)
               : "memory");
}

// ---------------------------------------------------------------------------
// f32 -> f16 weight conversion
// ---------------------------------------------------------------------------
__global__ __launch_bounds__(256) void cvt_kernel(const float* __restrict__ s,
                                                  _Float16* __restrict__ d, int n) {
  int i = blockIdx.x * 256 + threadIdx.x;
  if (i < n) d[i] = (_Float16)s[i];
}

// ---------------------------------------------------------------------------
// tb[b, j] = bqkv[j] + bt[j] + temb[b,:] @ Wt[j,:]   (tiny: 16 x 3456 dots)
// ---------------------------------------------------------------------------
__global__ __launch_bounds__(256) void temb_bias_kernel(
    const float* __restrict__ temb, const float* __restrict__ Wt,
    const float* __restrict__ bt, const float* __restrict__ bqkv,
    float* __restrict__ tb) {
  int j = blockIdx.x * 256 + threadIdx.x;
  int b = blockIdx.y;
  if (j >= C3) return;
  const float* t = temb + (size_t)b * CC;
  const float* w = Wt + (size_t)j * CC;
  float s = bqkv[j] + bt[j];
  for (int k = 0; k < CC; k++) s += t[k] * w[k];
  tb[(size_t)b * C3 + j] = s;
}

// ---------------------------------------------------------------------------
// RMSNorm (no affine), fp32 in -> f16 out. One block per row of length CC.
// ---------------------------------------------------------------------------
__global__ __launch_bounds__(256) void rmsnorm_kernel(const float* __restrict__ x,
                                                      _Float16* __restrict__ out) {
  const int row = blockIdx.x;
  const float* xr = x + (size_t)row * CC;
  float s = 0.f;
  for (int i = threadIdx.x; i < CC; i += 256) { float v = xr[i]; s += v * v; }
  __shared__ float red[256];
  red[threadIdx.x] = s;
  __syncthreads();
  for (int off = 128; off > 0; off >>= 1) {
    if (threadIdx.x < off) red[threadIdx.x] += red[threadIdx.x + off];
    __syncthreads();
  }
  const float r = rsqrtf(red[0] / (float)CC + 1e-6f);
  for (int i = threadIdx.x; i < CC; i += 256)
    out[(size_t)row * CC + i] = (_Float16)(xr[i] * r);
}

// ---------------------------------------------------------------------------
// WMMA GEMM with async-LDS double-buffered pipeline.
// out[M,Ntot] = A[M,K](f16) @ W[Ntot,K]^T(f16) + epilogue.
// Block = 256 thr = 8 waves (4Mx2N), block tile 128x64, K-step 32.
// Tiles are DMA'd global->LDS with GLOBAL_LOAD_ASYNC_TO_LDS_B128 (ASYNCcnt),
// double buffered: tile i+1 transfers while tile i feeds the WMMAs from LDS.
// Per-wave async instructions per tile = NPT, so s_wait_asynccnt <= NPT waits
// exactly for the current tile while the next stays in flight.
// MODE 0: + tb[batch(row), col]                       -> f32 out   (QKV)
// MODE 1: + bias[col] + resid[row,col]                -> f32 out   (proj / down)
// MODE 2: silu(acc+bg[col]) * (acc2+bu[col])          -> f16 out   (SwiGLU fused)
// ---------------------------------------------------------------------------
template <int MODE>
__global__ __launch_bounds__(256) void gemm_kernel(
    const _Float16* __restrict__ A, const _Float16* __restrict__ W0,
    const _Float16* __restrict__ W1, const float* __restrict__ bias0,
    const float* __restrict__ bias1, const float* __restrict__ resid,
    float* __restrict__ outf, _Float16* __restrict__ outh, int Ntot, int K) {
  const int tid = threadIdx.x;
  const int lane = tid & 31;
  const int wave = tid >> 5;
  const int m0b = blockIdx.x * 128;
  const int n0b = blockIdx.y * 64;
  const int wm = (wave >> 1) * 32;  // wave offsets inside the block tile
  const int wn = (wave & 1) * 32;
  const int r = lane & 15, hl = lane >> 4;
  constexpr int NPT = (MODE == 2) ? 4 : 3;  // async b128 per thread per tile

  __shared__ __align__(32) _Float16 Al[2][128][32];
  __shared__ __align__(32) _Float16 Bl[2][64][32];
  __shared__ __align__(32) _Float16 B2l[(MODE == 2) ? 2 : 1][64][32];

  v8f acc[2][2] = {};
  v8f acc2[2][2] = {};

  auto issue_tile = [&](int buf, int k0) {
    // A tile 128x32 f16 (8KB): 2 x b128 chunks per thread.
    int c = tid;
#pragma unroll
    for (int it = 0; it < 2; it++) {
      const int row = c >> 1, half = c & 1;
      async_g2l_b128(&Al[buf][row][half * 16],
                     A + (size_t)(m0b + row) * K + k0 + half * 16);
      c += 256;
    }
    // B tile 64x32 f16 (4KB): 1 x b128 per thread.
    const int row = tid >> 1, half = tid & 1;
    async_g2l_b128(&Bl[buf][row][half * 16],
                   W0 + (size_t)(n0b + row) * K + k0 + half * 16);
    if (MODE == 2)
      async_g2l_b128(&B2l[buf][row][half * 16],
                     W1 + (size_t)(n0b + row) * K + k0 + half * 16);
  };

  const int nk = K / 32;
  issue_tile(0, 0);
  for (int i = 0; i < nk; i++) {
    const int buf = i & 1;
    if (i + 1 < nk) {
      // Next tile's buffer was last read in iteration i-1 and fenced by that
      // iteration's trailing barrier -> safe to overwrite while we compute.
      issue_tile(buf ^ 1, (i + 1) * 32);
      asm volatile("s_wait_asynccnt %0" ::"i"(NPT) : "memory");
    } else {
      asm volatile("s_wait_asynccnt 0" ::: "memory");
    }
    __syncthreads();  // all waves' portions of tile i have landed in LDS

    const _Float16* Ab = &Al[buf][0][0];
    const _Float16* Bb = &Bl[buf][0][0];
    v16h a0 = load_a_frag(Ab, 32, wm, 0, lane);
    v16h a1 = load_a_frag(Ab, 32, wm + 16, 0, lane);
    v16h b0 = load_b_frag(Bb, 32, wn, 0, lane);
    v16h b1 = load_b_frag(Bb, 32, wn + 16, 0, lane);
    acc[0][0] = wmma16(a0, b0, acc[0][0]);
    acc[0][1] = wmma16(a0, b1, acc[0][1]);
    acc[1][0] = wmma16(a1, b0, acc[1][0]);
    acc[1][1] = wmma16(a1, b1, acc[1][1]);
    if (MODE == 2) {
      const _Float16* Cb = &B2l[buf][0][0];
      v16h c0 = load_b_frag(Cb, 32, wn, 0, lane);
      v16h c1 = load_b_frag(Cb, 32, wn + 16, 0, lane);
      acc2[0][0] = wmma16(a0, c0, acc2[0][0]);
      acc2[0][1] = wmma16(a0, c1, acc2[0][1]);
      acc2[1][0] = wmma16(a1, c0, acc2[1][0]);
      acc2[1][1] = wmma16(a1, c1, acc2[1][1]);
    }
    __syncthreads();  // tile i fully consumed before it can be overwritten
  }

#pragma unroll
  for (int mi = 0; mi < 2; mi++)
#pragma unroll
    for (int ni = 0; ni < 2; ni++)
#pragma unroll
      for (int j = 0; j < 8; j++) {
        const int row = m0b + wm + mi * 16 + hl * 8 + j;
        const int col = n0b + wn + ni * 16 + r;
        const float v = acc[mi][ni][j];
        if (MODE == 0) {
          outf[(size_t)row * Ntot + col] =
              v + bias0[(size_t)(row >> 10) * Ntot + col];
        } else if (MODE == 1) {
          outf[(size_t)row * Ntot + col] =
              v + bias0[col] + resid[(size_t)row * Ntot + col];
        } else {
          const float g = v + bias0[col];
          const float u = acc2[mi][ni][j] + bias1[col];
          const float sg = g / (1.0f + __expf(-g));
          outh[(size_t)row * Ntot + col] = (_Float16)(sg * u);
        }
      }
}

// ---------------------------------------------------------------------------
// QKV post: per-(b,h,n) wave does RMSNorm(q|k) w/ affine, axial RoPE,
// q *= 1/sqrt(hd), writes q/k as f16 rows padded to 96, v transposed [d][n]
// padded to 96 rows. RoPE pairs (i, i+18) per 36-wide half stay in-lane.
// ---------------------------------------------------------------------------
__global__ __launch_bounds__(256) void qkvpost_kernel(
    const float* __restrict__ qkv, const float* __restrict__ qnw,
    const float* __restrict__ knw, const float* __restrict__ cosy,
    const float* __restrict__ siny, const float* __restrict__ cosx,
    const float* __restrict__ sinx, _Float16* __restrict__ qh,
    _Float16* __restrict__ kh, _Float16* __restrict__ vt) {
  const int lane = threadIdx.x & 31;
  const int wid = blockIdx.x * 8 + (threadIdx.x >> 5);  // 0 .. B*H*N-1
  const int n = wid & (NN - 1);
  const int bh = wid >> 10;
  const int h = bh & (HH - 1);
  const int b = bh >> 4;

  const float* qp = qkv + (size_t)(b * NN + n) * C3 + h * HD;
  const float* kp = qp + CC;
  const float* vp = qp + 2 * CC;

  float q1[2], q2[2], k1[2], k2[2], v1[2], v2[2];
  float sq = 0.f, sk = 0.f;
#pragma unroll
  for (int t = 0; t < 2; t++) {
    const int p = lane + t * 32;
    if (p < 36) {
      const int quad = p / 18, i = p - quad * 18;
      const int d1 = quad * 36 + i, d2 = d1 + 18;
      q1[t] = qp[d1]; q2[t] = qp[d2];
      k1[t] = kp[d1]; k2[t] = kp[d2];
      v1[t] = vp[d1]; v2[t] = vp[d2];
      sq += q1[t] * q1[t] + q2[t] * q2[t];
      sk += k1[t] * k1[t] + k2[t] * k2[t];
    } else {
      q1[t] = q2[t] = k1[t] = k2[t] = v1[t] = v2[t] = 0.f;
    }
  }
#pragma unroll
  for (int m = 16; m > 0; m >>= 1) {
    sq += __shfl_xor(sq, m, 32);
    sk += __shfl_xor(sk, m, 32);
  }
  const float rq = rsqrtf(sq / (float)HD + 1e-6f);
  const float rk = rsqrtf(sk / (float)HD + 1e-6f);
  const float qs = rsqrtf((float)HD);  // fold 1/sqrt(hd) into q

  const size_t orow = ((size_t)bh * NN + n) * HDP;
#pragma unroll
  for (int t = 0; t < 2; t++) {
    const int p = lane + t * 32;
    if (p < 36) {
      const int quad = p / 18, i = p - quad * 18;
      const int d1 = quad * 36 + i, d2 = d1 + 18;
      const float* cs = quad ? cosx : cosy;
      const float* sn = quad ? sinx : siny;
      const float c1 = cs[n * 36 + i], s1 = sn[n * 36 + i];
      const float c2 = cs[n * 36 + i + 18], s2 = sn[n * 36 + i + 18];
      const float qa = q1[t] * rq * qnw[d1], qb = q2[t] * rq * qnw[d2];
      const float ka = k1[t] * rk * knw[d1], kb = k2[t] * rk * knw[d2];
      qh[orow + d1] = (_Float16)((qa * c1 - qb * s1) * qs);
      qh[orow + d2] = (_Float16)((qb * c2 + qa * s2) * qs);
      kh[orow + d1] = (_Float16)(ka * c1 - kb * s1);
      kh[orow + d2] = (_Float16)(kb * c2 + ka * s2);
      vt[((size_t)bh * HDP + d1) * NN + n] = (_Float16)v1[t];
      vt[((size_t)bh * HDP + d2) * NN + n] = (_Float16)v2[t];
    }
  }
  if (lane < HDP - HD) {  // zero padding d = 72..95
    qh[orow + HD + lane] = (_Float16)0.f;
    kh[orow + HD + lane] = (_Float16)0.f;
    vt[((size_t)bh * HDP + HD + lane) * NN + n] = (_Float16)0.f;
  }
}

// ---------------------------------------------------------------------------
// Flash attention, one (b,h) per blockIdx.y, 128 queries per block, 16 per wave.
// Per 32-key tile: 6 WMMAs for S = Q K^T, online softmax with half-wave
// shuffles, P staged through per-wave LDS (C->A layout), 6 WMMAs for O += P V.
// ---------------------------------------------------------------------------
__global__ __launch_bounds__(256) void attn_kernel(
    const _Float16* __restrict__ qh, const _Float16* __restrict__ kh,
    const _Float16* __restrict__ vt, _Float16* __restrict__ oh) {
  const int lane = threadIdx.x & 31;
  const int wave = threadIdx.x >> 5;
  const int r = lane & 15, hl = lane >> 4;
  const int bh = blockIdx.y;
  const int h = bh & (HH - 1), b = bh >> 4;
  const int qrow = blockIdx.x * 128 + wave * 16;

  const _Float16* Qp = qh + (size_t)bh * NN * HDP;
  const _Float16* Kp = kh + (size_t)bh * NN * HDP;
  const _Float16* Vp = vt + (size_t)bh * HDP * NN;

  v16h qa[3];
#pragma unroll
  for (int c = 0; c < 3; c++) qa[c] = load_a_frag(Qp, HDP, qrow, c * 32, lane);

  v8f o[6] = {};
  float rmax[8], rsum[8];
#pragma unroll
  for (int j = 0; j < 8; j++) { rmax[j] = -3.0e30f; rsum[j] = 0.f; }

  __shared__ __align__(32) _Float16 plds[8][16][32];

  for (int kt = 0; kt < NN; kt += 32) {
    v8f s0 = {}, s1 = {};
#pragma unroll
    for (int c = 0; c < 3; c++) {
      v16h kb0 = load_b_frag(Kp, HDP, kt, c * 32, lane);
      v16h kb1 = load_b_frag(Kp, HDP, kt + 16, c * 32, lane);
      s0 = wmma16(qa[c], kb0, s0);
      s1 = wmma16(qa[c], kb1, s1);
    }
#pragma unroll
    for (int j = 0; j < 8; j++) {
      float mx = fmaxf(s0[j], s1[j]);
#pragma unroll
      for (int msk = 8; msk > 0; msk >>= 1)
        mx = fmaxf(mx, __shfl_xor(mx, msk, 32));  // reduce within 16-lane half
      const float nm = fmaxf(rmax[j], mx);
      const float corr = __expf(rmax[j] - nm);
      const float p0 = __expf(s0[j] - nm);
      const float p1 = __expf(s1[j] - nm);
      float ps = p0 + p1;
#pragma unroll
      for (int msk = 8; msk > 0; msk >>= 1) ps += __shfl_xor(ps, msk, 32);
      rsum[j] = rsum[j] * corr + ps;
      rmax[j] = nm;
#pragma unroll
      for (int c = 0; c < 6; c++) o[c][j] *= corr;
      plds[wave][hl * 8 + j][r] = (_Float16)p0;  // C-layout row hl*8+j
      plds[wave][hl * 8 + j][r + 16] = (_Float16)p1;
    }
    // Re-read P as an A fragment (intra-wave: DS ops are in-order, no barrier).
    v16h pa;
    ((v8h*)&pa)[0] = *(const v8h*)&plds[wave][r][hl * 8];
    ((v8h*)&pa)[1] = *(const v8h*)&plds[wave][r][16 + hl * 8];
#pragma unroll
    for (int c = 0; c < 6; c++) {
      v16h vb = load_b_frag(Vp, NN, c * 16, kt, lane);
      o[c] = wmma16(pa, vb, o[c]);
    }
  }

#pragma unroll
  for (int c = 0; c < 6; c++) {
    const int d = c * 16 + r;
    if (d < HD) {
#pragma unroll
      for (int j = 0; j < 8; j++) {
        const int row = qrow + hl * 8 + j;
        oh[(size_t)(b * NN + row) * CC + h * HD + d] =
            (_Float16)(o[c][j] / rsum[j]);
      }
    }
  }
}

// ---------------------------------------------------------------------------
// Launch
// ---------------------------------------------------------------------------
extern "C" void kernel_launch(void* const* d_in, const int* in_sizes, int n_in,
                              void* d_out, int out_size, void* d_ws,
                              size_t ws_size, hipStream_t stream) {
  (void)in_sizes; (void)n_in; (void)out_size; (void)ws_size;
  const float* x    = (const float*)d_in[0];
  const float* temb = (const float*)d_in[1];
  const float* cosy = (const float*)d_in[2];
  const float* siny = (const float*)d_in[3];
  const float* cosx = (const float*)d_in[4];
  const float* sinx = (const float*)d_in[5];
  const float* Wqkv = (const float*)d_in[6];
  const float* bqkv = (const float*)d_in[7];
  const float* Wt   = (const float*)d_in[8];
  const float* bt   = (const float*)d_in[9];
  const float* Wp   = (const float*)d_in[10];
  const float* bp   = (const float*)d_in[11];
  const float* qnw  = (const float*)d_in[12];
  const float* knw  = (const float*)d_in[13];
  const float* Wg   = (const float*)d_in[14];
  const float* bg   = (const float*)d_in[15];
  const float* Wu   = (const float*)d_in[16];
  const float* bu   = (const float*)d_in[17];
  const float* Wd   = (const float*)d_in[18];
  const float* bd   = (const float*)d_in[19];
  float* out = (float*)d_out;

  // Workspace carve-up (all sizes multiples of 256B; qkv region reused for o/h).
  char* p = (char*)d_ws;
  const size_t SZ_XH  = (size_t)MM * CC * 2;   // 37.7 MB
  const size_t SZ_X1  = (size_t)MM * CC * 4;   // 75.5 MB
  const size_t SZ_QKV = (size_t)MM * C3 * 4;   // 226.5 MB (aliases oh + hbuf)
  const size_t SZ_TB  = (size_t)BB * C3 * 4;
  const size_t SZ_QH  = (size_t)BB * HH * NN * HDP * 2;  // 50.3 MB each
  _Float16* xh = (_Float16*)p;            p += SZ_XH;
  float*    x1 = (float*)p;               p += SZ_X1;
  char*     qr = p;                       p += SZ_QKV;
  float*    qkv  = (float*)qr;            // phase 1
  _Float16* ohb  = (_Float16*)qr;         // phase 2 (after qkv consumed)
  _Float16* hbuf = (_Float16*)(qr + SZ_XH);  // phase 3
  float*    tb = (float*)p;               p += SZ_TB;
  _Float16* qhb = (_Float16*)p;           p += SZ_QH;
  _Float16* khb = (_Float16*)p;           p += SZ_QH;
  _Float16* vtb = (_Float16*)p;           p += SZ_QH;
  _Float16* wqkvh = (_Float16*)p;         p += (size_t)C3 * CC * 2;
  _Float16* wph   = (_Float16*)p;         p += (size_t)CC * CC * 2;
  _Float16* wgh   = (_Float16*)p;         p += (size_t)HID * CC * 2;
  _Float16* wuh   = (_Float16*)p;         p += (size_t)HID * CC * 2;
  _Float16* wdh   = (_Float16*)p;         p += (size_t)CC * HID * 2;

  auto cvt = [&](const float* s, _Float16* d, int n) {
    cvt_kernel<<<(n + 255) / 256, 256, 0, stream>>>(s, d, n);
  };
  cvt(Wqkv, wqkvh, C3 * CC);
  cvt(Wp, wph, CC * CC);
  cvt(Wg, wgh, HID * CC);
  cvt(Wu, wuh, HID * CC);
  cvt(Wd, wdh, CC * HID);

  temb_bias_kernel<<<dim3((C3 + 255) / 256, BB), 256, 0, stream>>>(
      temb, Wt, bt, bqkv, tb);

  // norm1 -> f16
  rmsnorm_kernel<<<MM, 256, 0, stream>>>(x, xh);

  // qkv = xn @ Wqkv^T + tb[batch]   (M=16384, N=3456, K=1152)
  gemm_kernel<0><<<dim3(MM / 128, C3 / 64), 256, 0, stream>>>(
      xh, wqkvh, nullptr, tb, nullptr, nullptr, qkv, nullptr, C3, CC);

  // per-head RMS + RoPE + scale + layout (pads hd 72 -> 96, V transposed)
  qkvpost_kernel<<<(BB * HH * NN) / 8, 256, 0, stream>>>(
      qkv, qnw, knw, cosy, siny, cosx, sinx, qhb, khb, vtb);

  // flash attention: grid (1024/128 query tiles, 256 bh)
  attn_kernel<<<dim3(NN / 128, BB * HH), 256, 0, stream>>>(qhb, khb, vtb, ohb);

  // x1 = x + o @ Wp^T + bp
  gemm_kernel<1><<<dim3(MM / 128, CC / 64), 256, 0, stream>>>(
      ohb, wph, nullptr, bp, nullptr, x, x1, nullptr, CC, CC);

  // norm2 -> f16 (reuse xh)
  rmsnorm_kernel<<<MM, 256, 0, stream>>>(x1, xh);

  // h = silu(xn2 @ Wg^T + bg) * (xn2 @ Wu^T + bu)   fused dual-accumulator GEMM
  gemm_kernel<2><<<dim3(MM / 128, HID / 64), 256, 0, stream>>>(
      xh, wgh, wuh, bg, bu, nullptr, nullptr, hbuf, HID, CC);

  // out = x1 + h @ Wd^T + bd
  gemm_kernel<1><<<dim3(MM / 128, CC / 64), 256, 0, stream>>>(
      hbuf, wdh, nullptr, bd, nullptr, x1, out, nullptr, CC, HID);
}
